// LSTM_26164940767349
// MI455X (gfx1250) — matrix operand order repaired and checked by reading, not compiled, not run on hardware
//
#include <hip/hip_runtime.h>

// LSTM scan on gfx1250, v3:
//  - Phase 1 (parallel): Z = x_t @ Wx as bf16 WMMA GEMM, M=64 per WG
//    (4 batch tiles share every B fragment -> 4x less L2 traffic), 2048 WGs.
//  - Phase 2 (serial): recurrence with Wh split: gates f,i register-resident
//    (256 VGPRs/wave), gates c~,o in LDS (256 KB, loaded once) -> steady-state
//    loop touches L2 only for the 32KB/step Z stream.
//  - fallback fused kernel if workspace is too small for Z (256 MB).

namespace {
constexpr int kE     = 256;
constexpr int kT     = 512;
constexpr int kBT    = 16;    // batch rows per tile
constexpr int kPitch = 520;   // fused-kernel LDS pitch (shorts)
constexpr int kHP    = 264;   // x/h tile LDS pitch (shorts): 528B -> bank stride 4
}

typedef __attribute__((ext_vector_type(16))) __bf16   v16bf;
typedef __attribute__((ext_vector_type(8)))  __bf16   v8bf;
typedef __attribute__((ext_vector_type(8)))  float    v8f;
typedef __attribute__((ext_vector_type(8)))  _Float16 v8h;

__device__ __forceinline__ unsigned short f2bf(float f) {
  union { float f; unsigned int u; } c; c.f = f;
  unsigned int u = c.u;
  u += 0x7FFFu + ((u >> 16) & 1u);   // round-to-nearest-even
  return (unsigned short)(u >> 16);
}
__device__ __forceinline__ float sigmoid_fast(float x) {
  return 1.0f / (1.0f + __expf(-x));
}
__device__ __forceinline__ float tanh_fast(float x) {
  return 1.0f - 2.0f / (__expf(2.0f * x) + 1.0f);
}

// ---------------------------------------------------------------------------
// Pack Wf/Wi/Wc/Wo (fp32 [512][256]) into bf16 WMMA B-fragments.
// frag[g][ntile][ktile][lane][j]; ISA 16-bit B 32x16 layout:
// lane<16 -> K 0..7,16..23, lane>=16 -> K 8..15,24..31 (2/VGPR); N = lane&15.
// ktiles 0..7 = x-part rows of W, ktiles 8..15 = h-part rows.
// ---------------------------------------------------------------------------
__global__ __launch_bounds__(256)
void lstm_pack_weights(const float* __restrict__ Wf, const float* __restrict__ Wi,
                       const float* __restrict__ Wc, const float* __restrict__ Wo,
                       unsigned short* __restrict__ out)
{
  const int idx  = blockIdx.x * 256 + threadIdx.x;     // < 524288
  const int j    = idx & 15;
  const int lane = (idx >> 4) & 31;
  const int k    = (idx >> 9) & 15;
  const int n    = (idx >> 13) & 15;
  const int g    = idx >> 17;
  const float* W = (g == 0) ? Wf : (g == 1) ? Wi : (g == 2) ? Wc : Wo;
  const int K = k * 32 + ((j >> 3) << 4) + ((lane >> 4) << 3) + (j & 7);
  const int N = n * 16 + (lane & 15);
  out[idx] = f2bf(W[(size_t)K * kE + N]);
}

// ---------------------------------------------------------------------------
// Phase 1: Z[t][btile][combo][lane][8] (f16) = x_t @ Wx.
// One WG per (t, group of 4 btiles): M=64, each B fragment feeds 4 WMMAs.
// ---------------------------------------------------------------------------
__global__ __launch_bounds__(256)
void lstm_xpart_kernel(const float* __restrict__ X,
                       const unsigned short* __restrict__ WB,
                       unsigned short* __restrict__ Z)
{
  __shared__ unsigned short xb[4][kBT][kHP];    // 4 batch tiles (~34 KB)
  const int bi   = blockIdx.x;                  // t*4 + grp
  const int t    = bi >> 2;
  const int grp  = bi & 3;
  const int tid  = threadIdx.x;
  const int lane = tid & 31;
  const int wave = tid >> 5;
  const int ncol = lane & 15;
  const int hi   = lane >> 4;

  // stage 4 x-tiles (fp32 -> bf16) into LDS
  {
    const int xr = tid >> 4, xs = tid & 15;
    #pragma unroll
    for (int m = 0; m < 4; ++m) {
      const int b = (grp * 4 + m) * kBT + xr;
      const float4* src = (const float4*)(X + ((size_t)b * kT + t) * kE + xs * 16);
      float4 q0 = src[0], q1 = src[1], q2 = src[2], q3 = src[3];
      float v[16] = {q0.x, q0.y, q0.z, q0.w, q1.x, q1.y, q1.z, q1.w,
                     q2.x, q2.y, q2.z, q2.w, q3.x, q3.y, q3.z, q3.w};
      unsigned int* dst = (unsigned int*)&xb[m][xr][xs * 16];
      #pragma unroll
      for (int q = 0; q < 8; ++q)
        dst[q] = (unsigned int)f2bf(v[2 * q]) |
                 ((unsigned int)f2bf(v[2 * q + 1]) << 16);
    }
  }
  __syncthreads();

  const v8f vzero = {0.f, 0.f, 0.f, 0.f, 0.f, 0.f, 0.f, 0.f};
  v8f acc[4][8];
  #pragma unroll
  for (int m = 0; m < 4; ++m)
    #pragma unroll
    for (int c = 0; c < 8; ++c) acc[m][c] = vzero;

  const unsigned short* pB[8];
  #pragma unroll
  for (int c = 0; c < 8; ++c) {
    const int g = c >> 1, n = c & 1;
    pB[c] = WB + ((size_t)((g * 16 + (wave * 2 + n)) * 16) * 32 + lane) * 16;
  }

  #pragma unroll
  for (int k = 0; k < 8; ++k) {                 // x-part ktiles 0..7
    v16bf a[4];
    #pragma unroll
    for (int m = 0; m < 4; ++m) {
      const unsigned short* arow = &xb[m][ncol][hi * 8];
      const v8bf alo = *(const v8bf*)(arow + k * 32);
      const v8bf ahi = *(const v8bf*)(arow + k * 32 + 16);
      a[m] = __builtin_shufflevector(
          alo, ahi, 0, 1, 2, 3, 4, 5, 6, 7, 8, 9, 10, 11, 12, 13, 14, 15);
    }
    #pragma unroll
    for (int c = 0; c < 8; ++c) {
      const v16bf b = *(const v16bf*)(pB[c] + (size_t)k * 512);
      #pragma unroll
      for (int m = 0; m < 4; ++m)
        acc[m][c] = __builtin_amdgcn_wmma_f32_16x16x32_bf16(
            false, a[m], false, b, (short)0, acc[m][c], false, false);
    }
  }

  // store D tiles as f16 in lane-major Z layout
  #pragma unroll
  for (int m = 0; m < 4; ++m) {
    const int btile = grp * 4 + m;
    #pragma unroll
    for (int c = 0; c < 8; ++c) {
      const int g = c >> 1, n = c & 1;
      v8h z;
      #pragma unroll
      for (int j = 0; j < 8; ++j) z[j] = (_Float16)acc[m][c][j];
      *(v8h*)(Z + (((size_t)(t * 16 + btile) * 64 +
                    (g * 16 + wave * 2 + n)) * 32 + lane) * 8) = z;
    }
  }
}

// ---------------------------------------------------------------------------
// Phase 2 (serial): recurrence only. Wh split:
//   gates f,i  -> register-resident fragments (4 combos x 8 k = 256 VGPRs/wave)
//   gates c~,o -> LDS-resident fragments (256 KB, loaded from L2 once)
// h in LDS (bf16), c in VGPRs. Steady-state L2 traffic: Z stream only.
// ---------------------------------------------------------------------------
__global__ __launch_bounds__(256, 1)
void lstm_recur_kernel(const unsigned short* __restrict__ WB,
                       const unsigned short* __restrict__ Z,
                       const float* __restrict__ bF, const float* __restrict__ bI,
                       const float* __restrict__ bC, const float* __restrict__ bO,
                       float* __restrict__ out)
{
  __shared__ unsigned short hb[kBT][kHP];       // 8.4 KB h buffer
  __shared__ unsigned short wlds[2 * 16 * 8 * 512];  // 256 KB: [g-2][nt][k][lane][16]

  const int tid  = threadIdx.x;
  const int lane = tid & 31;
  const int wave = tid >> 5;
  const int b0   = blockIdx.x * kBT;
  const int ncol = lane & 15;
  const int hi   = lane >> 4;
  const int c0   = wave * 32;

  {  // h_{-1} = 0
    const int r = tid >> 4, s = (tid & 15) * 16;
    #pragma unroll
    for (int q = 0; q < 16; ++q) hb[r][s + q] = 0;
  }

  // preload gates 2..3 h-part fragments (ktiles 8..15) into LDS
  #pragma unroll 2
  for (int blk = 0; blk < 32; ++blk) {
    const int g = 2 + (blk >> 4), nt = blk & 15;
    const uint4* s = (const uint4*)(WB + (size_t)((g * 16 + nt) * 16 + 8) * 512);
    uint4* d = (uint4*)&wlds[blk * 4096];
    d[tid] = s[tid];
    d[tid + 256] = s[tid + 256];
  }

  float biasF[2], biasI[2], biasC[2], biasO[2];
  #pragma unroll
  for (int n = 0; n < 2; ++n) {
    const int col = c0 + n * 16 + ncol;
    biasF[n] = bF[col]; biasI[n] = bI[col];
    biasC[n] = bC[col]; biasO[n] = bO[col];
  }

  // register-resident Wh for gates 0..1: Whr[c][k], c = g*2+n (g in {0,1})
  v16bf Whr[4][8];
  #pragma unroll
  for (int c = 0; c < 4; ++c) {
    const int g = c >> 1, n = c & 1;
    const unsigned short* p =
        WB + ((size_t)((g * 16 + (wave * 2 + n)) * 16 + 8) * 32 + lane) * 16;
    #pragma unroll
    for (int k = 0; k < 8; ++k) Whr[c][k] = *(const v16bf*)(p + (size_t)k * 512);
  }

  // LDS fragment pointers for gates 2..3
  const unsigned short* wl[4];
  #pragma unroll
  for (int c = 0; c < 4; ++c) {
    const int blk = (c >> 1) * 16 + (wave * 2 + (c & 1));
    wl[c] = &wlds[blk * 4096 + lane * 16];
  }

  // Z pointers: one per combo (c = g*2+n, g 0..3), advanced by zstride each step
  const size_t zstride = (size_t)16 * 64 * 32 * 8;   // elements per t
  const unsigned short* zp[8];
  #pragma unroll
  for (int c = 0; c < 8; ++c) {
    const int g = c >> 1, n = c & 1;
    zp[c] = Z + (((size_t)blockIdx.x * 64 + (g * 16 + wave * 2 + n)) * 32 + lane) * 8;
  }

  const v8f vzero = {0.f, 0.f, 0.f, 0.f, 0.f, 0.f, 0.f, 0.f};
  v8f   cst[2] = {vzero, vzero};
  float hreg[2][8];
  const unsigned short* arow = &hb[ncol][hi * 8];

  __syncthreads();   // wlds + zeroed hb visible

  size_t zoff = 0;
  for (int t = 0; t < kT; ++t, zoff += zstride) {
    // acc init from precomputed x-part (f16 -> f32)
    v8f acc[8];
    #pragma unroll
    for (int c = 0; c < 8; ++c) {
      const v8h z = *(const v8h*)(zp[c] + zoff);
      #pragma unroll
      for (int j = 0; j < 8; ++j) acc[c][j] = (float)z[j];
    }
    if (t + 1 < kT) __builtin_prefetch(zp[0] + zoff + zstride, 0, 0);

    // h-part GEMM: K=256; gates 0..1 from registers, gates 2..3 from LDS
    #pragma unroll
    for (int k = 0; k < 8; ++k) {
      const v8bf alo = *(const v8bf*)(arow + k * 32);
      const v8bf ahi = *(const v8bf*)(arow + k * 32 + 16);
      const v16bf a = __builtin_shufflevector(
          alo, ahi, 0, 1, 2, 3, 4, 5, 6, 7, 8, 9, 10, 11, 12, 13, 14, 15);
      #pragma unroll
      for (int c = 0; c < 4; ++c)
        acc[c] = __builtin_amdgcn_wmma_f32_16x16x32_bf16(
            false, a, false, Whr[c][k], (short)0, acc[c], false, false);
      #pragma unroll
      for (int c = 0; c < 4; ++c) {
        const v16bf b = *(const v16bf*)(wl[c] + k * 512);
        acc[4 + c] = __builtin_amdgcn_wmma_f32_16x16x32_bf16(
            false, a, false, b, (short)0, acc[4 + c], false, false);
      }
    }

    // gate fusion: f=acc[0..1], i=acc[2..3], c~=acc[4..5], o=acc[6..7]
    #pragma unroll
    for (int n = 0; n < 2; ++n) {
      #pragma unroll
      for (int j = 0; j < 8; ++j) {
        const float fg = sigmoid_fast(acc[0 + n][j] + biasF[n]);
        const float ig = sigmoid_fast(acc[2 + n][j] + biasI[n]);
        const float cg = tanh_fast   (acc[4 + n][j] + biasC[n]);
        const float og = sigmoid_fast(acc[6 + n][j] + biasO[n]);
        const float cn = cst[n][j] * fg + ig * cg;
        cst[n][j] = cn;
        hreg[n][j] = og * tanh_fast(cn);
      }
    }

    __syncthreads();   // all waves done reading h_{t-1}

    #pragma unroll
    for (int n = 0; n < 2; ++n) {
      const int col = c0 + n * 16 + ncol;
      #pragma unroll
      for (int j = 0; j < 8; ++j)
        hb[j + hi * 8][col] = f2bf(hreg[n][j]);
    }
    __syncthreads();   // h_t visible before next step's reads
  }

  #pragma unroll
  for (int n = 0; n < 2; ++n)
    #pragma unroll
    for (int j = 0; j < 8; ++j)
      out[(size_t)(b0 + j + hi * 8) * kE + c0 + n * 16 + ncol] = hreg[n][j];
}

// ---------------------------------------------------------------------------
// Fallback: fused kernel, weights streamed from L2 every step.
// ---------------------------------------------------------------------------
__global__ __launch_bounds__(256, 1)
void lstm_fused_kernel(const float* __restrict__ X,
                       const unsigned short* __restrict__ WB,
                       const float* __restrict__ bF, const float* __restrict__ bI,
                       const float* __restrict__ bC, const float* __restrict__ bO,
                       float* __restrict__ out)
{
  __shared__ unsigned short xh[kBT][kPitch];
  const int tid  = threadIdx.x;
  const int lane = tid & 31;
  const int wave = tid >> 5;
  const int b0   = blockIdx.x * kBT;
  const int ncol = lane & 15;
  const int hi   = lane >> 4;
  const int c0   = wave * 32;

  {
    const int r = tid >> 4, s = (tid & 15) * 16;
    #pragma unroll
    for (int q = 0; q < 16; ++q) xh[r][kE + s + q] = 0;
  }

  float biasF[2], biasI[2], biasC[2], biasO[2];
  #pragma unroll
  for (int n = 0; n < 2; ++n) {
    const int col = c0 + n * 16 + ncol;
    biasF[n] = bF[col]; biasI[n] = bI[col];
    biasC[n] = bC[col]; biasO[n] = bO[col];
  }

  const unsigned short* pB[4][2];
  #pragma unroll
  for (int g = 0; g < 4; ++g)
    #pragma unroll
    for (int n = 0; n < 2; ++n)
      pB[g][n] = WB + ((size_t)((g * 16 + (wave * 2 + n)) * 16) * 32 + lane) * 16;

  const v8f vzero = {0.f, 0.f, 0.f, 0.f, 0.f, 0.f, 0.f, 0.f};
  v8f   cst[2] = {vzero, vzero};
  float hreg[2][8];

  const int xr = tid >> 4, xs = tid & 15;
  const float* xptr = X + ((size_t)(b0 + xr) * kT) * kE + xs * 16;
  unsigned int* xdst = (unsigned int*)&xh[xr][xs * 16];
  const unsigned short* arow = &xh[ncol][hi * 8];

  for (int t = 0; t < kT; ++t) {
    {
      const float4* src = (const float4*)xptr;
      float4 q0 = src[0], q1 = src[1], q2 = src[2], q3 = src[3];
      float v[16] = {q0.x, q0.y, q0.z, q0.w, q1.x, q1.y, q1.z, q1.w,
                     q2.x, q2.y, q2.z, q2.w, q3.x, q3.y, q3.z, q3.w};
      #pragma unroll
      for (int q = 0; q < 8; ++q)
        xdst[q] = (unsigned int)f2bf(v[2 * q]) |
                  ((unsigned int)f2bf(v[2 * q + 1]) << 16);
      if (t + 1 < kT) __builtin_prefetch(xptr + kE, 0, 0);
      xptr += kE;
    }
    __syncthreads();

    v8f acc[4][2];
    #pragma unroll
    for (int g = 0; g < 4; ++g)
      #pragma unroll
      for (int n = 0; n < 2; ++n) acc[g][n] = vzero;

    #pragma unroll 4
    for (int k = 0; k < 16; ++k) {
      const v8bf alo = *(const v8bf*)(arow + k * 32);
      const v8bf ahi = *(const v8bf*)(arow + k * 32 + 16);
      const v16bf a = __builtin_shufflevector(
          alo, ahi, 0, 1, 2, 3, 4, 5, 6, 7, 8, 9, 10, 11, 12, 13, 14, 15);
      #pragma unroll
      for (int g = 0; g < 4; ++g)
        #pragma unroll
        for (int n = 0; n < 2; ++n) {
          const v16bf b = *(const v16bf*)(pB[g][n] + (size_t)k * 512);
          acc[g][n] = __builtin_amdgcn_wmma_f32_16x16x32_bf16(
              false, a, false, b, (short)0, acc[g][n], false, false);
        }
    }

    #pragma unroll
    for (int n = 0; n < 2; ++n) {
      #pragma unroll
      for (int j = 0; j < 8; ++j) {
        const float fg = sigmoid_fast(acc[0][n][j] + biasF[n]);
        const float ig = sigmoid_fast(acc[1][n][j] + biasI[n]);
        const float cg = tanh_fast   (acc[2][n][j] + biasC[n]);
        const float og = sigmoid_fast(acc[3][n][j] + biasO[n]);
        const float cn = cst[n][j] * fg + ig * cg;
        cst[n][j] = cn;
        hreg[n][j] = og * tanh_fast(cn);
      }
    }

    __syncthreads();

    #pragma unroll
    for (int n = 0; n < 2; ++n) {
      const int col = kE + c0 + n * 16 + ncol;
      #pragma unroll
      for (int j = 0; j < 8; ++j)
        xh[j + hi * 8][col] = f2bf(hreg[n][j]);
    }
  }

  #pragma unroll
  for (int n = 0; n < 2; ++n)
    #pragma unroll
    for (int j = 0; j < 8; ++j)
      out[(size_t)(b0 + j + hi * 8) * kE + c0 + n * 16 + ncol] = hreg[n][j];
}

// ---------------------------------------------------------------------------
extern "C" void kernel_launch(void* const* d_in, const int* in_sizes, int n_in,
                              void* d_out, int out_size, void* d_ws, size_t ws_size,
                              hipStream_t stream) {
  (void)in_sizes; (void)n_in; (void)out_size;
  const float* X  = (const float*)d_in[0];
  const float* Wf = (const float*)d_in[1];
  const float* Wi = (const float*)d_in[2];
  const float* Wc = (const float*)d_in[3];
  const float* Wo = (const float*)d_in[4];
  const float* bf = (const float*)d_in[5];
  const float* bi = (const float*)d_in[6];
  const float* bc = (const float*)d_in[7];
  const float* bo = (const float*)d_in[8];

  const size_t wbBytes = (size_t)4 * 512 * 256 * 2;                  // 1 MB
  const size_t zBytes  = (size_t)512 * 16 * 64 * 32 * 8 * 2;          // 256 MB
  if (ws_size < wbBytes) return;

  unsigned short* WB = (unsigned short*)d_ws;
  lstm_pack_weights<<<2048, 256, 0, stream>>>(Wf, Wi, Wc, Wo, WB);

  if (ws_size >= wbBytes + zBytes) {
    unsigned short* Z = (unsigned short*)((char*)d_ws + wbBytes);
    lstm_xpart_kernel<<<512 * 4, 256, 0, stream>>>(X, WB, Z);
    lstm_recur_kernel<<<256 / kBT, 256, 0, stream>>>(
        WB, Z, bf, bi, bc, bo, (float*)d_out);
  } else {
    lstm_fused_kernel<<<256 / kBT, 256, 0, stream>>>(
        X, WB, bf, bi, bc, bo, (float*)d_out);
  }
}
